// GEKG_42949673563
// MI455X (gfx1250) — matrix operand order repaired and checked by reading, not compiled, not run on hardware
//
#include <hip/hip_runtime.h>

typedef __attribute__((ext_vector_type(16))) _Float16 v16h;
typedef __attribute__((ext_vector_type(8)))  float    v8f;

#define DIMV 32
#define KNBR 32
#define LHIST 50

__device__ __forceinline__ float sigmoidf(float x) {
  return 1.0f / (1.0f + __expf(-x));
}

// ---------------------------------------------------------------------------
// Prep kernel: convert W_gen [64,32] f32 -> fragment-major f16 in d_ws.
// Fragment id f = kb*2 + nt  (kb: K-half 0/1 i.e. ent/rel rows, nt: N-tile).
// For lane l (hk = l>>4, n = l&15), element j (0..15) of the v16h B-fragment
// is W_gen[kb*32 + (j<8 ? hk*8 + j : 16 + hk*8 + j-8)][nt*16 + n]   (B 32x16 KxN layout, wave32).
// Stored contiguously: ws[(f*32 + l)*16 + j], so the main kernel loads each
// fragment as one aligned 32-byte v16h (2x global_load_b128).
// ---------------------------------------------------------------------------
__global__ __launch_bounds__(128) void pack_wgen_f16(
    const float* __restrict__ Wg, _Float16* __restrict__ out) {
  int t    = threadIdx.x;       // 0..127 : (frag, lane)
  int frag = t >> 5;
  int lane = t & 31;
  int kb   = frag >> 1;
  int nt   = frag & 1;
  int hk   = lane >> 4;
  int n    = lane & 15;
#pragma unroll
  for (int j = 0; j < 16; ++j) {
    int krow = kb * 32 + (j < 8 ? hk * 8 + j : 16 + hk * 8 + (j - 8));
    out[(size_t)t * 16 + j] = (_Float16)Wg[krow * DIMV + nt * 16 + n];
  }
}

// ---------------------------------------------------------------------------
// Main kernel: one wave32 workgroup per batch element.
// ---------------------------------------------------------------------------
__global__ __launch_bounds__(32) void kgcn_item_user_kernel(
    const int*   __restrict__ user_records,  // [B,50]
    const int*   __restrict__ items,         // [B]
    const int*   __restrict__ nbr_e,         // [B,32]
    const int*   __restrict__ nbr_r,         // [B,32]
    const float* __restrict__ ent_emb,       // [1e6,32]
    const float* __restrict__ rel_emb,       // [64,32]
    const float* __restrict__ b_gen,         // [32]
    const float* __restrict__ W_att,         // [64]
    const float* __restrict__ b_att,         // [1]
    const _Float16* __restrict__ Wh,         // packed B fragments (4KB)
    float* __restrict__ out)                 // [B]
{
  __shared__ float s_ent[KNBR][DIMV + 1];   // +1 pad: conflict-free rows+cols
  __shared__ float s_rel[KNBR][DIMV + 1];
  __shared__ float s_gen[KNBR][DIMV + 1];
  __shared__ float s_watt[2 * DIMV];
  __shared__ float s_a[KNBR];

  const int b    = blockIdx.x;
  const int lane = threadIdx.x;             // 0..31
  const int hk   = lane >> 4;
  const int n    = lane & 15;

  // Stage W_att (broadcast-hot) into LDS once.
  s_watt[lane]        = W_att[lane];
  s_watt[lane + DIMV] = W_att[lane + DIMV];

  // Gather neighbor entity/relation rows: lane = dim, coalesced 128B rows.
  for (int k = 0; k < KNBR; ++k) {
    int ei = nbr_e[b * KNBR + k];
    int ri = nbr_r[b * KNBR + k];
    s_ent[k][lane] = ent_emb[(size_t)ei * DIMV + lane];
    s_rel[k][lane] = rel_emb[(size_t)ri * DIMV + lane];
  }

  // User embedding = sum of 50 gathered rows (lane = dim).
  float u = 0.0f;
  for (int i = 0; i < LHIST; ++i) {
    int ui = user_records[b * LHIST + i];
    u += ent_emb[(size_t)ui * DIMV + lane];
  }
  const float item_base = ent_emb[(size_t)items[b] * DIMV + lane];

  __syncthreads();

  // ---- attention pass 1 over raw e_r: lane = neighbor k -------------------
  const float batt = b_att[0];
  float s1 = batt;
#pragma unroll
  for (int d = 0; d < DIMV; ++d)
    s1 += s_ent[lane][d] * s_watt[d] + s_rel[lane][d] * s_watt[DIMV + d];
  float w1 = sigmoidf(s1);

  float mx = w1;
#pragma unroll
  for (int o = 16; o >= 1; o >>= 1) mx = fmaxf(mx, __shfl_xor(mx, o, 32));
  float e1 = __expf(w1 - mx);
  float sm = e1;
#pragma unroll
  for (int o = 16; o >= 1; o >>= 1) sm += __shfl_xor(sm, o, 32);
  s_a[lane] = e1 / sm;
  __syncthreads();

  float nbr = 0.0f;                          // lane = dim
#pragma unroll
  for (int k = 0; k < KNBR; ++k) nbr += s_ent[k][lane] * s_a[k];

  // ---- gen = sigmoid(e_r @ W_gen + b_gen) via v_wmma_f32_16x16x32_f16 -----
  // A fragments (16x32 MxK f16): mt = M-tile (neighbors), kb=0 ent / kb=1 rel.
  v16h afr[2][2];
#pragma unroll
  for (int mt = 0; mt < 2; ++mt) {
    int row = mt * 16 + n;                   // n == lane&15 == M within tile
#pragma unroll
    for (int j = 0; j < 8; ++j) {
      afr[mt][0][j]     = (_Float16)s_ent[row][hk * 8 + j];
      afr[mt][0][j + 8] = (_Float16)s_ent[row][16 + hk * 8 + j];
      afr[mt][1][j]     = (_Float16)s_rel[row][hk * 8 + j];
      afr[mt][1][j + 8] = (_Float16)s_rel[row][16 + hk * 8 + j];
    }
  }
  // B fragments: 32-byte aligned contiguous loads from packed scratch.
  v16h bfr[2][2];
#pragma unroll
  for (int kb = 0; kb < 2; ++kb)
#pragma unroll
    for (int nt = 0; nt < 2; ++nt)
      bfr[kb][nt] = *(const v16h*)(Wh + ((size_t)(kb * 2 + nt) * 32 + lane) * 16);

#pragma unroll
  for (int mt = 0; mt < 2; ++mt) {
#pragma unroll
    for (int nt = 0; nt < 2; ++nt) {
      v8f c = {};
      c = __builtin_amdgcn_wmma_f32_16x16x32_f16(
              false, afr[mt][0], false, bfr[0][nt], (short)0, c, false, false);
      c = __builtin_amdgcn_wmma_f32_16x16x32_f16(
              false, afr[mt][1], false, bfr[1][nt], (short)0, c, false, false);
      float bg = b_gen[nt * 16 + n];
#pragma unroll
      for (int r = 0; r < 8; ++r)            // D: M = hk*8 + r, N = n
        s_gen[mt * 16 + hk * 8 + r][nt * 16 + n] = sigmoidf(c[r] + bg);
    }
  }
  __syncthreads();

  // ---- attention pass 2 over [gen | rel]: lane = neighbor k ---------------
  float s2 = batt;
#pragma unroll
  for (int d = 0; d < DIMV; ++d)
    s2 += s_gen[lane][d] * s_watt[d] + s_rel[lane][d] * s_watt[DIMV + d];
  float w2 = sigmoidf(s2);

  mx = w2;
#pragma unroll
  for (int o = 16; o >= 1; o >>= 1) mx = fmaxf(mx, __shfl_xor(mx, o, 32));
  float e2 = __expf(w2 - mx);
  sm = e2;
#pragma unroll
  for (int o = 16; o >= 1; o >>= 1) sm += __shfl_xor(sm, o, 32);
  __syncthreads();                           // retire pass-1 reads of s_a
  s_a[lane] = e2 / sm;
  __syncthreads();

  float gnbr = 0.0f;                         // lane = dim
#pragma unroll
  for (int k = 0; k < KNBR; ++k) gnbr += s_gen[k][lane] * s_a[k];

  // ---- final score --------------------------------------------------------
  float item_e = item_base + nbr + gnbr;
  float p = u * item_e;
#pragma unroll
  for (int o = 16; o >= 1; o >>= 1) p += __shfl_xor(p, o, 32);
  if (lane == 0) out[b] = sigmoidf(p);
}

extern "C" void kernel_launch(void* const* d_in, const int* in_sizes, int n_in,
                              void* d_out, int out_size, void* d_ws, size_t ws_size,
                              hipStream_t stream) {
  const int*   user_records = (const int*)d_in[0];
  const int*   items        = (const int*)d_in[1];
  const int*   nbr_e        = (const int*)d_in[2];
  const int*   nbr_r        = (const int*)d_in[3];
  const float* ent_emb      = (const float*)d_in[4];
  const float* rel_emb      = (const float*)d_in[5];
  const float* W_gen        = (const float*)d_in[6];
  const float* b_gen        = (const float*)d_in[7];
  const float* W_att        = (const float*)d_in[8];
  const float* b_att        = (const float*)d_in[9];
  float*       out          = (float*)d_out;
  _Float16*    Wh           = (_Float16*)d_ws;   // 4 KB of packed B fragments

  const int B = in_sizes[1];                      // items is [B]

  pack_wgen_f16<<<1, 128, 0, stream>>>(W_gen, Wh);
  kgcn_item_user_kernel<<<B, 32, 0, stream>>>(
      user_records, items, nbr_e, nbr_r, ent_emb, rel_emb,
      b_gen, W_att, b_att, Wh, out);
}
